// Sequence_85564338471528
// MI455X (gfx1250) — matrix operand order repaired and checked by reading, not compiled, not run on hardware
//
#include <hip/hip_runtime.h>
#include <hip/hip_bf16.h>
#include <math.h>

#define Bsz  128
#define Tlen 512
#define Hd   512
#define G4   2048
#define PITCH 520   // LDS row pitch in bf16 elems: 512 + 8 pad (breaks bank conflicts)

typedef __attribute__((ext_vector_type(16))) __bf16 v16bf;
typedef __attribute__((ext_vector_type(8)))  __bf16 v8bf;
typedef __attribute__((ext_vector_type(8)))  float  v8f;

__device__ __forceinline__ unsigned short f2bf(float f) {
    unsigned u = __float_as_uint(f);
    u += 0x7fffu + ((u >> 16) & 1u);          // round-to-nearest-even
    return (unsigned short)(u >> 16);
}
__device__ __forceinline__ float sigm(float x) { return 1.0f / (1.0f + __expf(-x)); }

// acc[4] += A(16x32 rows of arow) * B(32x16 from LDS weight slice), over K=512
__device__ __forceinline__ void gemm_acc(v8f* acc, const unsigned short* __restrict__ arow,
                                         const unsigned short* sW, int m, int hf) {
#pragma unroll 4
    for (int kc = 0; kc < Hd; kc += 32) {
        v8bf alo = *reinterpret_cast<const v8bf*>(arow + kc + hf * 8);
        v8bf ahi = *reinterpret_cast<const v8bf*>(arow + kc + 16 + hf * 8);
        v16bf a = __builtin_shufflevector(alo, ahi, 0,1,2,3,4,5,6,7,8,9,10,11,12,13,14,15);
#pragma unroll
        for (int g = 0; g < 4; ++g) {
            const unsigned short* brow = sW + (g * 16 + m) * PITCH + kc + hf * 8;
            v8bf blo = *reinterpret_cast<const v8bf*>(brow);
            v8bf bhi = *reinterpret_cast<const v8bf*>(brow + 16);
            v16bf b = __builtin_shufflevector(blo, bhi, 0,1,2,3,4,5,6,7,8,9,10,11,12,13,14,15);
            acc[g] = __builtin_amdgcn_wmma_f32_16x16x32_bf16(
                false, a, false, b, (short)0, acc[g], false, false);
        }
    }
}

// ---------------- prep kernels ----------------
__global__ void k_init(unsigned short* hbase, unsigned* bar, float* out, const float* b2) {
    int i = blockIdx.x * blockDim.x + threadIdx.x;
    int stride = gridDim.x * blockDim.x;
    if (i == 0) { bar[0] = 0; bar[1] = 0; }
    for (int j = i; j < 6 * Bsz * Hd; j += stride) hbase[j] = 0;
    float bv = b2[0];
    for (int j = i; j < Bsz * Tlen; j += stride) out[j] = bv;
}

__global__ void k_cvt(const float* __restrict__ src, unsigned short* __restrict__ dst, int n) {
    int i = blockIdx.x * blockDim.x + threadIdx.x;
    int stride = gridDim.x * blockDim.x;
    for (int j = i; j < n; j += stride) dst[j] = f2bf(src[j]);
}

// v1[g] = Wih1[g,:]·W1col ; c1[g] = Wih1[g,:]·b1 + bih1+bhh1 ; c2/c3 = bias sums
__global__ void k_vconst(const float* Wih1, const float* W1, const float* b1,
                         const float* bih1, const float* bhh1,
                         const float* bih2, const float* bhh2,
                         const float* bih3, const float* bhh3,
                         float* v1, float* c1, float* c2, float* c3) {
    int g = blockIdx.x * blockDim.x + threadIdx.x;
    if (g >= G4) return;
    float sv = 0.f, sc = 0.f;
    for (int h = 0; h < Hd; ++h) {
        float w = Wih1[g * Hd + h];
        sv += w * W1[h];
        sc += w * b1[h];
    }
    v1[g] = sv;
    c1[g] = sc + bih1[g] + bhh1[g];
    c2[g] = bih2[g] + bhh2[g];
    c3[g] = bih3[g] + bhh3[g];
}

// ---------------- main persistent pipelined kernel ----------------
// grid = 96 blocks (layer = blockIdx/32, col-slice = blockIdx%32), 256 threads (8 waves)
__global__ __launch_bounds__(256) void k_lstm(
    const float* __restrict__ x, const float* __restrict__ W2,
    const float* __restrict__ v1, const float* __restrict__ cst1,
    const float* __restrict__ cst2, const float* __restrict__ cst3,
    const unsigned short* __restrict__ Whh1, const unsigned short* __restrict__ Wih2,
    const unsigned short* __restrict__ Whh2, const unsigned short* __restrict__ Wih3,
    const unsigned short* __restrict__ Whh3,
    unsigned short* h1, unsigned short* h2, unsigned short* h3,   // each 2*B*H (double buffered)
    float* out, unsigned* bar)
{
    extern __shared__ unsigned short smem[];           // 2 * 64 * PITCH bf16
    unsigned short* s_rec = smem;
    unsigned short* s_in  = smem + 64 * PITCH;

    const int layer = blockIdx.x >> 5;
    const int wgc   = blockIdx.x & 31;
    const int wave  = threadIdx.x >> 5;
    const int lane  = threadIdx.x & 31;
    const int hf    = lane >> 4;         // lane half
    const int m     = lane & 15;
    const int rbase = wave << 4;         // batch-row tile base
    const int cbase = wgc << 4;          // hidden-col tile base

    const unsigned short* Wrec = (layer == 0) ? Whh1 : (layer == 1) ? Whh2 : Whh3;
    const unsigned short* Winp = (layer == 1) ? Wih2 : Wih3;
    unsigned short* hosel = (layer == 0) ? h1 : (layer == 1) ? h2 : h3;
    const unsigned short* hisel = (layer == 1) ? h1 : h2;

    // stage this WG's weight slices (64 gate-rows x 512 K) into LDS, padded pitch
    for (int idx = threadIdx.x; idx < 64 * 64; idx += 256) {
        int row = idx >> 6, ch = (idx & 63) << 3;
        int g = row >> 4, n = row & 15;
        v8bf v = *reinterpret_cast<const v8bf*>(Wrec + ((size_t)(g * Hd + cbase + n)) * Hd + ch);
        *reinterpret_cast<v8bf*>(s_rec + row * PITCH + ch) = v;
    }
    if (layer > 0) {
        for (int idx = threadIdx.x; idx < 64 * 64; idx += 256) {
            int row = idx >> 6, ch = (idx & 63) << 3;
            int g = row >> 4, n = row & 15;
            v8bf v = *reinterpret_cast<const v8bf*>(Winp + ((size_t)(g * Hd + cbase + n)) * Hd + ch);
            *reinterpret_cast<v8bf*>(s_in + row * PITCH + ch) = v;
        }
    }
    __syncthreads();

    const float* cstp = (layer == 0) ? cst1 : (layer == 1) ? cst2 : cst3;
    float cst[4], vg[4];
#pragma unroll
    for (int g = 0; g < 4; ++g) {
        cst[g] = cstp[g * Hd + cbase + m];
        vg[g]  = (layer == 0) ? v1[g * Hd + cbase + m] : 0.0f;
    }
    const float w2v = W2[cbase + m];

    v8f cstate;
#pragma unroll
    for (int r = 0; r < 8; ++r) cstate[r] = 0.0f;

    for (int tick = 0; tick < Tlen + 2; ++tick) {
        const int t = tick - layer;
        if (t >= 0 && t < Tlen) {
            const int cur = tick & 1;
            const unsigned short* hrec = hosel + (cur ^ 1) * (Bsz * Hd);
            unsigned short*       hout = hosel + cur * (Bsz * Hd);
            const unsigned short* hin  = hisel + (cur ^ 1) * (Bsz * Hd);

            v8f acc[4];
#pragma unroll
            for (int g = 0; g < 4; ++g)
#pragma unroll
                for (int r = 0; r < 8; ++r) acc[g][r] = cst[g];
            if (layer == 0) {        // rank-1 input contribution: x_t * (Wih1 @ W1col)
#pragma unroll
                for (int r = 0; r < 8; ++r) {
                    float xv = x[(rbase + r + 8 * hf) * Tlen + t];
#pragma unroll
                    for (int g = 0; g < 4; ++g) acc[g][r] += xv * vg[g];
                }
            }

            const unsigned short* arow = hrec + (rbase + m) * Hd;
            gemm_acc(acc, arow, s_rec, m, hf);
            if (layer > 0) {
                const unsigned short* arow2 = hin + (rbase + m) * Hd;
                gemm_acc(acc, arow2, s_in, m, hf);
            }

            // LSTM cell update (c kept in registers across all timesteps)
#pragma unroll
            for (int r = 0; r < 8; ++r) {
                float iv = sigm(acc[0][r]);
                float fv = sigm(acc[1][r]);
                float gv = tanhf(acc[2][r]);
                float ov = sigm(acc[3][r]);
                float cn = fv * cstate[r] + iv * gv;
                cstate[r] = cn;
                float hv = ov * tanhf(cn);
                hout[(rbase + r + 8 * hf) * Hd + cbase + m] = f2bf(hv);
                if (layer == 2) {     // fused output projection h3 @ W2^T (+b2 preloaded)
                    float s = hv * w2v;
                    s += __shfl_xor(s, 1, 16);
                    s += __shfl_xor(s, 2, 16);
                    s += __shfl_xor(s, 4, 16);
                    s += __shfl_xor(s, 8, 16);
                    if (m == 0) atomicAdd(&out[(rbase + r + 8 * hf) * Tlen + t], s);
                }
            }
        }

        // one grid barrier per tick (layer-pipelined, so 1 sync instead of 3)
        __syncthreads();
        if (threadIdx.x == 0) {
            volatile unsigned* vgen = bar + 1;
            unsigned g = *vgen;
            __threadfence();
            unsigned prev = atomicAdd(bar, 1u);
            if (prev == (unsigned)(gridDim.x - 1)) {
                bar[0] = 0;
                __threadfence();
                atomicAdd((unsigned*)(bar + 1), 1u);
            } else {
                while (*vgen == g) __builtin_amdgcn_s_sleep(1);
            }
            __threadfence();
        }
        __syncthreads();
    }
}

extern "C" void kernel_launch(void* const* d_in, const int* in_sizes, int n_in,
                              void* d_out, int out_size, void* d_ws, size_t ws_size,
                              hipStream_t stream) {
    (void)in_sizes; (void)n_in; (void)out_size; (void)ws_size;
    const float* x    = (const float*)d_in[0];
    const float* W1   = (const float*)d_in[2];
    const float* b1   = (const float*)d_in[3];
    const float* W2   = (const float*)d_in[4];
    const float* b2   = (const float*)d_in[5];
    const float* Wih1 = (const float*)d_in[6];
    const float* Whh1 = (const float*)d_in[7];
    const float* bih1 = (const float*)d_in[8];
    const float* bhh1 = (const float*)d_in[9];
    const float* Wih2 = (const float*)d_in[10];
    const float* Whh2 = (const float*)d_in[11];
    const float* bih2 = (const float*)d_in[12];
    const float* bhh2 = (const float*)d_in[13];
    const float* Wih3 = (const float*)d_in[14];
    const float* Whh3 = (const float*)d_in[15];
    const float* bih3 = (const float*)d_in[16];
    const float* bhh3 = (const float*)d_in[17];

    char* ws = (char*)d_ws;
    size_t off = 0;
    auto take = [&](size_t bytes) -> char* {
        char* p = ws + off;
        off = (off + bytes + 255) & ~(size_t)255;
        return p;
    };
    unsigned* bar = (unsigned*)take(256);
    float* v1  = (float*)take(G4 * 4);
    float* c1v = (float*)take(G4 * 4);
    float* c2v = (float*)take(G4 * 4);
    float* c3v = (float*)take(G4 * 4);
    const size_t wbytes = (size_t)G4 * Hd * 2;
    unsigned short* bWhh1 = (unsigned short*)take(wbytes);
    unsigned short* bWih2 = (unsigned short*)take(wbytes);
    unsigned short* bWhh2 = (unsigned short*)take(wbytes);
    unsigned short* bWih3 = (unsigned short*)take(wbytes);
    unsigned short* bWhh3 = (unsigned short*)take(wbytes);
    unsigned short* hbase = (unsigned short*)take((size_t)6 * Bsz * Hd * 2);
    unsigned short* h1 = hbase;
    unsigned short* h2 = hbase + 2 * Bsz * Hd;
    unsigned short* h3 = hbase + 4 * Bsz * Hd;

    k_init<<<256, 256, 0, stream>>>(hbase, bar, (float*)d_out, b2);
    k_cvt<<<512, 256, 0, stream>>>(Whh1, bWhh1, G4 * Hd);
    k_cvt<<<512, 256, 0, stream>>>(Wih2, bWih2, G4 * Hd);
    k_cvt<<<512, 256, 0, stream>>>(Whh2, bWhh2, G4 * Hd);
    k_cvt<<<512, 256, 0, stream>>>(Wih3, bWih3, G4 * Hd);
    k_cvt<<<512, 256, 0, stream>>>(Whh3, bWhh3, G4 * Hd);
    k_vconst<<<8, 256, 0, stream>>>(Wih1, W1, b1, bih1, bhh1, bih2, bhh2, bih3, bhh3,
                                    v1, c1v, c2v, c3v);

    const int shmem = 2 * 64 * PITCH * 2;   // 133120 B of LDS (well under 320 KB/WGP)
    hipFuncSetAttribute(reinterpret_cast<const void*>(k_lstm),
                        hipFuncAttributeMaxDynamicSharedMemorySize, shmem);
    k_lstm<<<96, 256, shmem, stream>>>(x, W2, v1, c1v, c2v, c3v,
                                       bWhh1, bWih2, bWhh2, bWih3, bWhh3,
                                       h1, h2, h3, (float*)d_out, bar);
}
